// UCBAttention_37340445671724
// MI455X (gfx1250) — matrix operand order repaired and checked by reading, not compiled, not run on hardware
//
#include <hip/hip_runtime.h>

typedef unsigned short u16;
typedef unsigned int   u32;
typedef unsigned long long u64;

typedef __attribute__((__ext_vector_type__(16))) __bf16 v16bf;
typedef __attribute__((__ext_vector_type__(8)))  float  v8f;

#define B_   16
#define N_   577
#define C_   1024
#define H_   16
#define DH   64
#define M_   (B_ * N_)     /* 9232 */
#define NPAD 608           /* 577 padded to multiple of 32 */
#define KK_  144           /* top-k keys kept */
#define SSTR 612           /* LDS score-row stride (floats), breaks bank conflicts */

union FragB { v16bf v; u16 u[16]; uint4 q[2]; };
union AccU  { v8f  v; float f[8]; };

__device__ __forceinline__ u16 f2bf(float x) {
  u32 u = __float_as_uint(x);
  u32 r = u + 0x7FFFu + ((u >> 16) & 1u);
  return (u16)(r >> 16);
}
__device__ __forceinline__ int wred_add(int v) {
  #pragma unroll
  for (int o = 16; o > 0; o >>= 1) v += __shfl_xor(v, o, 32);
  return v;
}
__device__ __forceinline__ float wred_addf(float v) {
  #pragma unroll
  for (int o = 16; o > 0; o >>= 1) v += __shfl_xor(v, o, 32);
  return v;
}
__device__ __forceinline__ float wred_maxf(float v) {
  #pragma unroll
  for (int o = 16; o > 0; o >>= 1) v = fmaxf(v, __shfl_xor(v, o, 32));
  return v;
}

/* ---- CDNA5 async global->LDS copies (ASYNCcnt tracked) ------------------ */
/* INST_OFFSET is added to BOTH the LDS and global addresses (ISA 08 §4.4), */
/* so one address pair moves a contiguous 32B run with two b128 ops.        */
__device__ __forceinline__ void async_b128x2(u32 lds_off, const void* gptr) {
  u64 ga = (u64)(uintptr_t)gptr;
  asm volatile("global_load_async_to_lds_b128 %0, %1, off\n\t"
               "global_load_async_to_lds_b128 %0, %1, off offset:16"
               :: "v"(lds_off), "v"(ga) : "memory");
}
__device__ __forceinline__ void async_b128(u32 lds_off, const void* gptr) {
  u64 ga = (u64)(uintptr_t)gptr;
  asm volatile("global_load_async_to_lds_b128 %0, %1, off"
               :: "v"(lds_off), "v"(ga) : "memory");
}
__device__ __forceinline__ void async_wait3() { asm volatile("s_wait_asynccnt 0x3" ::: "memory"); }
__device__ __forceinline__ void async_wait0() { asm volatile("s_wait_asynccnt 0x0" ::: "memory"); }
/* low 32 bits of a flat LDS pointer ARE the LDS byte offset (aperture in hi) */
__device__ __forceinline__ u32 lds_off32(const void* p) { return (u32)(uintptr_t)p; }

/* ---------------- init: converts, expl, pad-zeroing, delta-zeroing ------- */
#define T0 ((long)M_ * C_)          /* x -> bf16 */
#define T1 ((long)3 * C_ * C_)      /* qkv_w -> bf16 */
#define T2 ((long)C_ * C_)          /* proj_w -> bf16 */
#define T3 ((long)H_ * N_)          /* expl */
#define T4 ((long)B_ * H_ * 31 * DH)/* k pad rows zero */
#define T5 ((long)B_ * H_ * DH * 31)/* vT pad cols zero */
#define T6 ((long)H_ * N_)          /* delta zero */
#define TTOT (T0 + T1 + T2 + T3 + T4 + T5 + T6)

__launch_bounds__(256)
__global__ void init_kernel(const float* __restrict__ x, const float* __restrict__ qkvw,
                            const float* __restrict__ projw, const float* __restrict__ cnt,
                            const int* __restrict__ counterp,
                            u16* __restrict__ xbf, u16* __restrict__ wqkv, u16* __restrict__ wproj,
                            float* __restrict__ expl, u16* __restrict__ kws, u16* __restrict__ vtws,
                            int* __restrict__ gdelta) {
  long stride = (long)gridDim.x * 256;
  for (long i = (long)blockIdx.x * 256 + threadIdx.x; i < TTOT; i += stride) {
    long r = i;
    if (r < T0) { xbf[r] = f2bf(x[r]); continue; }
    r -= T0;
    if (r < T1) { wqkv[r] = f2bf(qkvw[r]); continue; }
    r -= T1;
    if (r < T2) { wproj[r] = f2bf(projw[r]); continue; }
    r -= T2;
    if (r < T3) {
      float lc = __logf((float)counterp[0] + 1.0f);
      expl[r] = __fsqrt_rn(lc / (cnt[r] + 1e-6f));
      continue;
    }
    r -= T3;
    if (r < T4) { /* zero k pad rows 577..607 */
      int bh = (int)(r / (31 * DH)); int rem = (int)(r % (31 * DH));
      int row = 577 + rem / DH; int d = rem % DH;
      kws[((size_t)bh * NPAD + row) * DH + d] = 0; continue;
    }
    r -= T4;
    if (r < T5) { /* zero vT pad cols 577..607 */
      int bh = (int)(r / (DH * 31)); int rem = (int)(r % (DH * 31));
      int d = rem / 31; int c = rem % 31;
      vtws[((size_t)bh * DH + d) * NPAD + 577 + c] = 0; continue;
    }
    r -= T5;
    gdelta[r] = 0;
  }
}

/* ---------------- QKV GEMM: (9232x1024) x (1024x3072)^T, WMMA bf16 ------ */
/* double-buffered LDS, async global->LDS staging                           */
__launch_bounds__(256)
__global__ void gemm_qkv_kernel(const u16* __restrict__ xbf, const u16* __restrict__ wqkv,
                                u16* __restrict__ qws, u16* __restrict__ kws,
                                u16* __restrict__ vtws) {
  __shared__ u16 lA[2][128 * 40];
  __shared__ u16 lB[2][64 * 40];
  const int tid = threadIdx.x, lane = tid & 31, wave = tid >> 5;
  const int m0 = blockIdx.x * 128, n0 = blockIdx.y * 64;
  const int wm = wave >> 1, wn = wave & 1;
  AccU acc[2][2];
  #pragma unroll
  for (int i = 0; i < 2; ++i)
    #pragma unroll
    for (int j = 0; j < 2; ++j)
      #pragma unroll
      for (int q = 0; q < 8; ++q) acc[i][j].f[q] = 0.f;

  const int rowA = tid >> 1, halfA = tid & 1;
  const int rowB = tid >> 2, qB = tid & 3;
  const int mA = (m0 + rowA < M_) ? (m0 + rowA) : (M_ - 1); /* clamp: rows >=M_ never stored */
  const u16* gA = xbf + (size_t)mA * C_ + halfA * 16;
  const u16* gB = wqkv + (size_t)(n0 + rowB) * C_ + qB * 8;
  const u32 dA0 = lds_off32(&lA[0][rowA * 40 + halfA * 16]);
  const u32 dA1 = lds_off32(&lA[1][rowA * 40 + halfA * 16]);
  const u32 dB0 = lds_off32(&lB[0][rowB * 40 + qB * 8]);
  const u32 dB1 = lds_off32(&lB[1][rowB * 40 + qB * 8]);

  /* prologue: stage k-step 0 into buffer 0 */
  async_b128x2(dA0, gA);
  async_b128(dB0, gB);

  for (int it = 0; it < 32; ++it) {
    const int cur = it & 1;
    if (it + 1 < 32) { /* stream next tile into the other buffer */
      const int kb = (it + 1) * 32;
      async_b128x2(cur ? dA0 : dA1, gA + kb);
      async_b128(cur ? dB0 : dB1, gB + kb);
      async_wait3();   /* current stage (3 ops) complete; next still in flight */
    } else {
      async_wait0();
    }
    __syncthreads();
    const int r15 = lane & 15, hf = lane >> 4;
    FragB a0, a1, b0, b1;
    { const u16* p = &lA[cur][(wm * 32 + r15) * 40 + hf * 8];
      a0.q[0] = *(const uint4*)p; a0.q[1] = *(const uint4*)(p + 16);
      p += 16 * 40;
      a1.q[0] = *(const uint4*)p; a1.q[1] = *(const uint4*)(p + 16); }
    { const u16* p = &lB[cur][(wn * 32 + r15) * 40 + hf * 8];
      b0.q[0] = *(const uint4*)p; b0.q[1] = *(const uint4*)(p + 16);
      p += 16 * 40;
      b1.q[0] = *(const uint4*)p; b1.q[1] = *(const uint4*)(p + 16); }
    acc[0][0].v = __builtin_amdgcn_wmma_f32_16x16x32_bf16(false, a0.v, false, b0.v, (short)0, acc[0][0].v, false, false);
    acc[0][1].v = __builtin_amdgcn_wmma_f32_16x16x32_bf16(false, a0.v, false, b1.v, (short)0, acc[0][1].v, false, false);
    acc[1][0].v = __builtin_amdgcn_wmma_f32_16x16x32_bf16(false, a1.v, false, b0.v, (short)0, acc[1][0].v, false, false);
    acc[1][1].v = __builtin_amdgcn_wmma_f32_16x16x32_bf16(false, a1.v, false, b1.v, (short)0, acc[1][1].v, false, false);
    __syncthreads();  /* all reads of buf[cur] done before it is restaged */
  }
  /* epilogue: scatter to q [key][d], k [key][d], vT [d][key] (all bf16) */
  const int r15 = lane & 15, mOff = (lane >> 4) * 8;
  #pragma unroll
  for (int i = 0; i < 2; ++i)
    #pragma unroll
    for (int j = 0; j < 2; ++j) {
      int col = n0 + wn * 32 + j * 16 + r15;
      int t = col >> 10, hh = (col >> 6) & 15, d = col & 63;
      #pragma unroll
      for (int q = 0; q < 8; ++q) {
        int m = m0 + wm * 32 + i * 16 + mOff + q;
        if (m >= M_) continue;
        int bIdx = m / N_, nn = m - bIdx * N_;
        int bh = bIdx * H_ + hh;
        u16 v = f2bf(acc[i][j].f[q]);
        if (t == 0)      qws[((size_t)bh * NPAD + nn) * DH + d] = v;
        else if (t == 1) kws[((size_t)bh * NPAD + nn) * DH + d] = v;
        else             vtws[((size_t)bh * DH + d) * NPAD + nn] = v;
      }
    }
}

/* ---------------- Attention core: scores + UCB top-k + softmax + context - */
__launch_bounds__(256)
__global__ void attn_kernel(const u16* __restrict__ qws, const u16* __restrict__ kws,
                            const u16* __restrict__ vtws, const float* __restrict__ expl,
                            u16* __restrict__ ctx, int* __restrict__ gdelta,
                            const int* __restrict__ counterp, const int* __restrict__ ucbp) {
  __shared__ float sS[32 * SSTR];
  __shared__ u16   sQ[32 * 72];
  __shared__ float sExpl[NPAD];
  __shared__ float sInv[32];
  __shared__ int   sCnt[NPAD];

  const int tid = threadIdx.x, lane = tid & 31, wave = tid >> 5;
  const int bh = blockIdx.y;
  const int h = bh & 15;
  const int bIdx = bh >> 4;
  const int q0 = blockIdx.x * 32;
  const bool ucbOn = (ucbp[0] != 0) && (counterp[0] > 50);

  /* stage query tile (async to LDS) + expl row, zero counters */
  { int row = tid >> 3, part = tid & 7;
    async_b128(lds_off32(&sQ[row * 72 + part * 8]),
               qws + ((size_t)bh * NPAD + q0 + row) * DH + part * 8); }
  for (int k = tid; k < NPAD; k += 256) {
    sExpl[k] = (k < N_) ? expl[h * N_ + k] : 0.f;
    sCnt[k] = 0;
  }
  async_wait0();
  __syncthreads();

  /* phase 1: S = scale * Q @ K^T into sS (fp32) */
  for (int t = wave; t < 76; t += 8) {
    int qs = t / 38, kt = t % 38;
    const int r15 = lane & 15, hf = lane >> 4;
    AccU acc;
    #pragma unroll
    for (int q = 0; q < 8; ++q) acc.f[q] = 0.f;
    #pragma unroll
    for (int kc = 0; kc < 2; ++kc) {
      FragB a, bfr;
      const u16* pq = &sQ[(qs * 16 + r15) * 72 + kc * 32 + hf * 8];
      a.q[0] = *(const uint4*)pq; a.q[1] = *(const uint4*)(pq + 16);
      int key = kt * 16 + r15;
      const u16* pk = kws + ((size_t)bh * NPAD + key) * DH + kc * 32 + hf * 8;
      bfr.q[0] = *(const uint4*)pk; bfr.q[1] = *(const uint4*)(pk + 16);
      acc.v = __builtin_amdgcn_wmma_f32_16x16x32_bf16(false, a.v, false, bfr.v, (short)0, acc.v, false, false);
    }
    int col = kt * 16 + r15, mOff = (lane >> 4) * 8;
    #pragma unroll
    for (int q = 0; q < 8; ++q)
      sS[(qs * 16 + mOff + q) * SSTR + col] = acc.f[q] * 0.125f;
  }
  __syncthreads();

  /* phase 2: per-row exact top-144 on (S + expl), masked softmax, counts */
  for (int rr = 0; rr < 4; ++rr) {
    int row = wave * 4 + rr;
    int qg = q0 + row;
    if (qg >= N_) {
      for (int k = lane; k < SSTR; k += 32) sS[row * SSTR + k] = 0.f;
      if (lane == 0) sInv[row] = 0.f;
      continue;
    }
    u32 keyu[18]; float sval[18]; u32 allow = 0;
    #pragma unroll
    for (int j = 0; j < 18; ++j) {
      int k = 1 + lane + 32 * j;
      float s = sS[row * SSTR + k];
      sval[j] = s;
      u32 b = __float_as_uint(s + sExpl[k]);
      keyu[j] = b ^ ((b >> 31) ? 0xFFFFFFFFu : 0x80000000u);
    }
    if (ucbOn) {
      u32 ans = 0;
      for (int bit = 31; bit >= 0; --bit) {
        u32 cand = ans | (1u << bit);
        int c = 0;
        #pragma unroll
        for (int j = 0; j < 18; ++j) c += (keyu[j] >= cand) ? 1 : 0;
        if (wred_add(c) >= KK_) ans = cand;
      }
      int gt = 0;
      #pragma unroll
      for (int j = 0; j < 18; ++j) gt += (keyu[j] > ans) ? 1 : 0;
      gt = wred_add(gt);
      int need = KK_ - gt;
      int tieBase = 0;
      u32 below_mask = (lane == 0) ? 0u : ((1u << lane) - 1u);
      #pragma unroll
      for (int j = 0; j < 18; ++j) {
        bool g = (keyu[j] > ans);
        bool te = (keyu[j] == ans);
        u32 bm = __builtin_amdgcn_ballot_w32(te);
        int below = __popc(bm & below_mask);
        bool sel = te && ((tieBase + below) < need);
        tieBase += __popc(bm);
        if (g || sel) allow |= (1u << j);
      }
    } else {
      allow = 0x3FFFFu;
    }
    float m = sS[row * SSTR + 0];
    #pragma unroll
    for (int j = 0; j < 18; ++j)
      if (allow & (1u << j)) m = fmaxf(m, sval[j]);
    m = wred_maxf(m);
    float part = 0.f;
    #pragma unroll
    for (int j = 0; j < 18; ++j) {
      int k = 1 + lane + 32 * j;
      float e = 0.f;
      if (allow & (1u << j)) { e = __expf(sval[j] - m); part += e; }
      sS[row * SSTR + k] = e;
    }
    if (lane == 0) {
      float e0 = __expf(sS[row * SSTR + 0] - m);
      sS[row * SSTR + 0] = e0;
      part += e0;
    }
    for (int k = N_ + lane; k < SSTR; k += 32) sS[row * SSTR + k] = 0.f;
    float tot = wred_addf(part);
    if (lane == 0) sInv[row] = 1.0f / tot;
    if (ucbOn) {
      if (lane == 0) atomicAdd(&sCnt[0], 1);
      #pragma unroll
      for (int j = 0; j < 18; ++j)
        if (allow & (1u << j)) atomicAdd(&sCnt[1 + lane + 32 * j], 1);
    }
  }
  __syncthreads();
  if (ucbOn) {
    for (int k = tid; k < N_; k += 256) {
      int c = sCnt[k];
      if (c) atomicAdd(&gdelta[h * N_ + k], c);
    }
  }
  __syncthreads();

  /* phase 3: context = P @ V (on-the-fly fp32->bf16 A fragments) */
  {
    const int qs = wave >> 2, ds = wave & 3;
    const int r15 = lane & 15, hf = lane >> 4;
    const int row = qs * 16 + r15;
    const float inv = sInv[row];
    AccU acc;
    #pragma unroll
    for (int q = 0; q < 8; ++q) acc.f[q] = 0.f;
    for (int kt = 0; kt < 19; ++kt) {
      const float* ps = &sS[row * SSTR + kt * 32 + hf * 8];
      FragB a;
      #pragma unroll
      for (int j = 0; j < 8; ++j) {
        a.u[j]     = f2bf(ps[j] * inv);
        a.u[j + 8] = f2bf(ps[j + 16] * inv);
      }
      const u16* pv = vtws + ((size_t)bh * DH + ds * 16 + r15) * NPAD + kt * 32 + hf * 8;
      FragB bfr;
      bfr.q[0] = *(const uint4*)pv; bfr.q[1] = *(const uint4*)(pv + 16);
      acc.v = __builtin_amdgcn_wmma_f32_16x16x32_bf16(false, a.v, false, bfr.v, (short)0, acc.v, false, false);
    }
    int mOff = hf * 8;
    #pragma unroll
    for (int q = 0; q < 8; ++q) {
      int qg = q0 + qs * 16 + mOff + q;
      if (qg < N_)
        ctx[((size_t)(bIdx * N_ + qg)) * C_ + h * DH + ds * 16 + r15] = f2bf(acc.f[q]);
    }
  }
}

/* ---------------- Proj GEMM: (9232x1024) x (1024x1024)^T + bias ---------- */
__launch_bounds__(256)
__global__ void gemm_proj_kernel(const u16* __restrict__ ctx, const u16* __restrict__ wproj,
                                 const float* __restrict__ bias, float* __restrict__ out) {
  __shared__ u16 lA[2][128 * 40];
  __shared__ u16 lB[2][64 * 40];
  const int tid = threadIdx.x, lane = tid & 31, wave = tid >> 5;
  const int m0 = blockIdx.x * 128, n0 = blockIdx.y * 64;
  const int wm = wave >> 1, wn = wave & 1;
  AccU acc[2][2];
  #pragma unroll
  for (int i = 0; i < 2; ++i)
    #pragma unroll
    for (int j = 0; j < 2; ++j)
      #pragma unroll
      for (int q = 0; q < 8; ++q) acc[i][j].f[q] = 0.f;

  const int rowA = tid >> 1, halfA = tid & 1;
  const int rowB = tid >> 2, qB = tid & 3;
  const int mA = (m0 + rowA < M_) ? (m0 + rowA) : (M_ - 1);
  const u16* gA = ctx + (size_t)mA * C_ + halfA * 16;
  const u16* gB = wproj + (size_t)(n0 + rowB) * C_ + qB * 8;
  const u32 dA0 = lds_off32(&lA[0][rowA * 40 + halfA * 16]);
  const u32 dA1 = lds_off32(&lA[1][rowA * 40 + halfA * 16]);
  const u32 dB0 = lds_off32(&lB[0][rowB * 40 + qB * 8]);
  const u32 dB1 = lds_off32(&lB[1][rowB * 40 + qB * 8]);

  async_b128x2(dA0, gA);
  async_b128(dB0, gB);

  for (int it = 0; it < 32; ++it) {
    const int cur = it & 1;
    if (it + 1 < 32) {
      const int kb = (it + 1) * 32;
      async_b128x2(cur ? dA0 : dA1, gA + kb);
      async_b128(cur ? dB0 : dB1, gB + kb);
      async_wait3();
    } else {
      async_wait0();
    }
    __syncthreads();
    const int r15 = lane & 15, hf = lane >> 4;
    FragB a0, a1, b0, b1;
    { const u16* p = &lA[cur][(wm * 32 + r15) * 40 + hf * 8];
      a0.q[0] = *(const uint4*)p; a0.q[1] = *(const uint4*)(p + 16);
      p += 16 * 40;
      a1.q[0] = *(const uint4*)p; a1.q[1] = *(const uint4*)(p + 16); }
    { const u16* p = &lB[cur][(wn * 32 + r15) * 40 + hf * 8];
      b0.q[0] = *(const uint4*)p; b0.q[1] = *(const uint4*)(p + 16);
      p += 16 * 40;
      b1.q[0] = *(const uint4*)p; b1.q[1] = *(const uint4*)(p + 16); }
    acc[0][0].v = __builtin_amdgcn_wmma_f32_16x16x32_bf16(false, a0.v, false, b0.v, (short)0, acc[0][0].v, false, false);
    acc[0][1].v = __builtin_amdgcn_wmma_f32_16x16x32_bf16(false, a0.v, false, b1.v, (short)0, acc[0][1].v, false, false);
    acc[1][0].v = __builtin_amdgcn_wmma_f32_16x16x32_bf16(false, a1.v, false, b0.v, (short)0, acc[1][0].v, false, false);
    acc[1][1].v = __builtin_amdgcn_wmma_f32_16x16x32_bf16(false, a1.v, false, b1.v, (short)0, acc[1][1].v, false, false);
    __syncthreads();
  }
  const int r15 = lane & 15, mOff = (lane >> 4) * 8;
  #pragma unroll
  for (int i = 0; i < 2; ++i)
    #pragma unroll
    for (int j = 0; j < 2; ++j) {
      int col = n0 + wn * 32 + j * 16 + r15;
      float bv = bias[col];
      #pragma unroll
      for (int q = 0; q < 8; ++q) {
        int m = m0 + wm * 32 + i * 16 + mOff + q;
        if (m < M_) out[(size_t)m * C_ + col] = acc[i][j].f[q] + bv;
      }
    }
}

/* ------------------------------- launcher -------------------------------- */
extern "C" void kernel_launch(void* const* d_in, const int* in_sizes, int n_in,
                              void* d_out, int out_size, void* d_ws, size_t ws_size,
                              hipStream_t stream) {
  const float* x        = (const float*)d_in[0];
  const float* qkvw     = (const float*)d_in[1];
  const float* projw    = (const float*)d_in[2];
  const float* projb    = (const float*)d_in[3];
  const float* cnt      = (const float*)d_in[4];
  const int*   counterp = (const int*)d_in[5];
  const int*   ucbp     = (const int*)d_in[6];

  char* ws = (char*)d_ws;
  size_t o = 0;
  auto alloc = [&](size_t bytes) -> void* {
    void* p = ws + o;
    o += (bytes + 255) & ~(size_t)255;
    return p;
  };
  u16*   xbf   = (u16*)alloc((size_t)M_ * C_ * 2);
  u16*   wqkv  = (u16*)alloc((size_t)3 * C_ * C_ * 2);
  u16*   wproj = (u16*)alloc((size_t)C_ * C_ * 2);
  u16*   qws   = (u16*)alloc((size_t)B_ * H_ * NPAD * DH * 2);
  u16*   kws   = (u16*)alloc((size_t)B_ * H_ * NPAD * DH * 2);
  u16*   vtws  = (u16*)alloc((size_t)B_ * H_ * DH * NPAD * 2);
  u16*   ctx   = (u16*)alloc((size_t)M_ * C_ * 2);
  float* expl  = (float*)alloc((size_t)H_ * N_ * 4);

  float* outp   = (float*)d_out;
  int*   gdelta = (int*)d_out + (size_t)M_ * C_;

  init_kernel<<<2048, 256, 0, stream>>>(x, qkvw, projw, cnt, counterp,
                                        xbf, wqkv, wproj, expl, kws, vtws, gdelta);
  gemm_qkv_kernel<<<dim3(73, 48), 256, 0, stream>>>(xbf, wqkv, qws, kws, vtws);
  attn_kernel<<<dim3(19, B_ * H_), 256, 0, stream>>>(qws, kws, vtws, expl, ctx, gdelta,
                                                     counterp, ucbp);
  gemm_proj_kernel<<<dim3(73, 16), 256, 0, stream>>>(ctx, wproj, projb, outp);
}
